// Seq2Seq_7095285973236
// MI455X (gfx1250) — matrix operand order repaired and checked by reading, not compiled, not run on hardware
//
#include <hip/hip_runtime.h>
#include <hip/hip_bf16.h>

// ---------------- problem constants ----------------
constexpr int kH    = 512;    // hidden
constexpr int kG    = 2048;   // 4*H
constexpr int kB    = 512;    // batch
constexpr int kT    = 128;    // encoder timesteps
constexpr int kHor  = 24;     // decoder horizon
constexpr int kNWG  = 64;     // persistent workgroups (2 mblk x 32 hblk)
constexpr int kThr  = 256;    // threads per WG (8 wave32)

// LDS weight tiles (per WG): 64 gate-cols x K, +8 bf16 row pad for banking.
constexpr int kW0Stride = kH + 8;        // 520 elems
constexpr int kW1Stride = 2 * kH + 8;    // 1032 elems
constexpr int kLdsElems = 64 * kW0Stride + 64 * kW1Stride;   // 99,328 bf16
constexpr int kLdsBytes = kLdsElems * 2;                     // 198,656 B (< 320KB/WGP)

typedef __attribute__((ext_vector_type(16))) __bf16 v16bf;
typedef __attribute__((ext_vector_type(8)))  __bf16 v8bf;
typedef __attribute__((ext_vector_type(8)))  float  v8f;

__device__ __forceinline__ float sigmoidf_(float x) {
  return 1.0f / (1.0f + __expf(-x));
}

__device__ __forceinline__ v16bf cat8(v8bf lo, v8bf hi) {
  return __builtin_shufflevector(lo, hi, 0,1,2,3,4,5,6,7,8,9,10,11,12,13,14,15);
}

// Device-wide barrier: monotonically increasing counter, one arrival per WG.
__device__ __forceinline__ void grid_bar(unsigned* cnt, unsigned& gen) {
  __syncthreads();
  __threadfence();                       // release: publish our stores (L2)
  ++gen;
  if (threadIdx.x == 0) {
    atomicAdd(cnt, 1u);
    const unsigned target = gen * (unsigned)kNWG;
    while (__hip_atomic_load(cnt, __ATOMIC_RELAXED, __HIP_MEMORY_SCOPE_AGENT) < target) {
      __builtin_amdgcn_s_sleep(1);
    }
  }
  __syncthreads();
  __threadfence();                       // acquire: invalidate WGP-local caches
}

// Copy one 64-row weight tile (rows = {t*512 + u0 + i}) into LDS, row-major + pad.
__device__ __forceinline__ void stage_tile(__bf16* dst, int dstride,
                                           const __bf16* __restrict__ src,
                                           int K, int u0, int tid) {
  const int chunksPerRow = K / 8;
  for (int id = tid; id < 64 * chunksPerRow; id += kThr) {
    const int r = id / chunksPerRow;
    const int c = (id - r * chunksPerRow) * 8;
    const int grow = (r >> 4) * kH + u0 + (r & 15);   // gate t = r>>4, unit i = r&15
    *(v8bf*)(dst + r * dstride + c) = *(const v8bf*)(src + (size_t)grow * K + c);
  }
}

// Two-row-block GEMM: each B fragment (from LDS) feeds two WMMAs (rows rowA and
// rowA+16), halving LDS traffic per MAC. Fragment layouts per CDNA5 ISA 7.12.2.
__device__ __forceinline__ void gemm_k2(v8f accA[4], v8f accB[4],
    const __bf16* __restrict__ A, int lda, int rowA,
    const __bf16* Wlds, int ws, int kofs, int kl, int lane) {
  const int m  = lane & 15;
  const int hi = lane >> 4;
  for (int kb = 0; kb < kl; kb += 32) {
    const __bf16* pa0 = A + (size_t)(rowA + m) * lda + kb + hi * 8;
    const __bf16* pa1 = pa0 + (size_t)16 * lda;
    v16bf a0 = cat8(*(const v8bf*)pa0, *(const v8bf*)(pa0 + 16));
    v16bf a1 = cat8(*(const v8bf*)pa1, *(const v8bf*)(pa1 + 16));
#pragma unroll
    for (int t = 0; t < 4; ++t) {
      const __bf16* pw = Wlds + (t * 16 + m) * ws + kofs + kb + hi * 8;
      v16bf b = cat8(*(const v8bf*)pw, *(const v8bf*)(pw + 16));
      accA[t] = __builtin_amdgcn_wmma_f32_16x16x32_bf16(
                  false, a0, false, b, (short)0, accA[t], false, false);
      accB[t] = __builtin_amdgcn_wmma_f32_16x16x32_bf16(
                  false, a1, false, b, (short)0, accB[t], false, false);
    }
  }
}

// f32 -> bf16 weight repack. Wcat = [Wih1 ; Whh1] concatenated along K (ld=1024).
__global__ void cvt_weights(
    const float* __restrict__ eWhh0, const float* __restrict__ eWih1, const float* __restrict__ eWhh1,
    const float* __restrict__ dWhh0, const float* __restrict__ dWih1, const float* __restrict__ dWhh1,
    __bf16* __restrict__ oWhh0e, __bf16* __restrict__ oCat1e,
    __bf16* __restrict__ oWhh0d, __bf16* __restrict__ oCat1d) {
  int i = blockIdx.x * blockDim.x + threadIdx.x;
  if (i >= kG * kH) return;
  int r = i >> 9;         // row in [0,2048)
  int k = i & (kH - 1);   // k in [0,512)
  oWhh0e[i] = (__bf16)eWhh0[i];
  oWhh0d[i] = (__bf16)dWhh0[i];
  oCat1e[(size_t)r * (2 * kH) + k]       = (__bf16)eWih1[i];
  oCat1e[(size_t)r * (2 * kH) + kH + k]  = (__bf16)eWhh1[i];
  oCat1d[(size_t)r * (2 * kH) + k]       = (__bf16)dWih1[i];
  oCat1d[(size_t)r * (2 * kH) + kH + k]  = (__bf16)dWhh1[i];
}

// One persistent kernel: 128 encoder steps then 24 decoder steps.
// c1/c2 live in registers the whole time; h1/h2 ping-pong in global (L2);
// recurrent weights live in LDS (staged once per phase).
__global__ void __launch_bounds__(kThr)
seq2seq_fused(const float* __restrict__ x,
    const float* __restrict__ eWih0, const float* __restrict__ eBi0, const float* __restrict__ eBh0,
    const float* __restrict__ eBi1,  const float* __restrict__ eBh1,
    const float* __restrict__ dWih0, const float* __restrict__ dBi0, const float* __restrict__ dBh0,
    const float* __restrict__ dBi1,  const float* __restrict__ dBh1,
    const float* __restrict__ fcW,   const float* __restrict__ fcb,
    const __bf16* __restrict__ Whh0e, const __bf16* __restrict__ Wcat1e,
    const __bf16* __restrict__ Whh0d, const __bf16* __restrict__ Wcat1d,
    __bf16* __restrict__ h1buf, __bf16* __restrict__ h2buf,
    float* __restrict__ xnext, unsigned* cnt, float* __restrict__ out) {
  extern __shared__ __align__(16) __bf16 smem[];
  __bf16* ldsW0 = smem;                      // 64 x kW0Stride
  __bf16* ldsW1 = smem + 64 * kW0Stride;     // 64 x kW1Stride

  const int tid  = threadIdx.x;
  const int lane = tid & 31;
  const int wave = tid >> 5;
  const int wg   = blockIdx.x;
  const int hblk = wg & 31;          // 32 hidden blocks of 16 units
  const int mblk = wg >> 5;          // 2 batch blocks of 256 rows
  const int u0   = hblk * 16;
  const int r0   = mblk * 256;
  const int rowA = r0 + wave * 32;                       // first of 32 rows (2 frags)
  const int n    = lane & 15;
  const int u    = u0 + n;                               // hidden unit this thread owns
  const int rowC0 = rowA + ((lane >> 4) << 3);           // first of 8 C rows (group 0)

  float c1[16], c2[16];
#pragma unroll
  for (int r = 0; r < 16; ++r) { c1[r] = 0.0f; c2[r] = 0.0f; }

  unsigned gen = 0;
  float bias1[4], bias2[4], wi0[4], wi1[4];

  for (int st = 0; st < kT + kHor; ++st) {
    const bool dec = (st >= kT);
    if (st == 0 || st == kT) {
      const float* Wih = dec ? dWih0 : eWih0;
      const float* bi0 = dec ? dBi0  : eBi0;
      const float* bh0 = dec ? dBh0  : eBh0;
      const float* bi1 = dec ? dBi1  : eBi1;
      const float* bh1 = dec ? dBh1  : eBh1;
#pragma unroll
      for (int t = 0; t < 4; ++t) {
        int j = t * kH + u;
        bias1[t] = bi0[j] + bh0[j];
        bias2[t] = bi1[j] + bh1[j];
        wi0[t] = Wih[j * 2 + 0];
        wi1[t] = Wih[j * 2 + 1];
      }
      stage_tile(ldsW0, kW0Stride, dec ? Whh0d : Whh0e, kH,     u0, tid);
      stage_tile(ldsW1, kW1Stride, dec ? Wcat1d : Wcat1e, 2 * kH, u0, tid);
      __syncthreads();
    }

    const int p = st & 1;
    const __bf16* h1r = h1buf + (size_t)p        * kB * kH;
    __bf16*       h1w = h1buf + (size_t)(p ^ 1)  * kB * kH;
    const __bf16* h2r = h2buf + (size_t)p        * kB * kH;
    __bf16*       h2w = h2buf + (size_t)(p ^ 1)  * kB * kH;

    // ---------------- layer 0: gates = h1_prev @ Whh0^T (+ x-proj + biases)
    {
      v8f acc[2][4] = {};
      gemm_k2(acc[0], acc[1], h1r, kH, rowA, ldsW0, kW0Stride, 0, kH, lane);
#pragma unroll
      for (int g = 0; g < 2; ++g) {
        const int rbase = rowC0 + g * 16;
        float* cg = c1 + g * 8;
        float hnew[8];
#pragma unroll
        for (int r = 0; r < 8; ++r) {
          const int b = rbase + r;
          float x0, x1;
          if (!dec) {
            const float* xp = x + ((size_t)b * kT + st) * 2;   // x[b, t, :]
            x0 = xp[0]; x1 = xp[1];
          } else {
            x0 = xnext[2 * b]; x1 = xnext[2 * b + 1];
          }
          float gi = acc[g][0][r] + bias1[0] + x0 * wi0[0] + x1 * wi1[0];
          float gf = acc[g][1][r] + bias1[1] + x0 * wi0[1] + x1 * wi1[1];
          float gg = acc[g][2][r] + bias1[2] + x0 * wi0[2] + x1 * wi1[2];
          float go = acc[g][3][r] + bias1[3] + x0 * wi0[3] + x1 * wi1[3];
          cg[r] = sigmoidf_(gf) * cg[r] + sigmoidf_(gi) * tanhf(gg);
          hnew[r] = sigmoidf_(go) * tanhf(cg[r]);
        }
#pragma unroll
        for (int r = 0; r < 8; ++r)
          h1w[(size_t)(rbase + r) * kH + u] = (__bf16)hnew[r];
      }
    }
    grid_bar(cnt, gen);

    // ---------------- layer 1: gates = [h1_cur, h2_prev] @ [Wih1;Whh1]^T + biases
    {
      v8f acc[2][4] = {};
      gemm_k2(acc[0], acc[1], h1w, kH, rowA, ldsW1, kW1Stride, 0,  kH, lane);
      gemm_k2(acc[0], acc[1], h2r, kH, rowA, ldsW1, kW1Stride, kH, kH, lane);
#pragma unroll
      for (int g = 0; g < 2; ++g) {
        const int rbase = rowC0 + g * 16;
        float* cg = c2 + g * 8;
        float hnew[8];
#pragma unroll
        for (int r = 0; r < 8; ++r) {
          float gi = acc[g][0][r] + bias2[0];
          float gf = acc[g][1][r] + bias2[1];
          float gg = acc[g][2][r] + bias2[2];
          float go = acc[g][3][r] + bias2[3];
          cg[r] = sigmoidf_(gf) * cg[r] + sigmoidf_(gi) * tanhf(gg);
          hnew[r] = sigmoidf_(go) * tanhf(cg[r]);
        }
#pragma unroll
        for (int r = 0; r < 8; ++r)
          h2w[(size_t)(rbase + r) * kH + u] = (__bf16)hnew[r];
      }
    }
    grid_bar(cnt, gen);

    // ---------------- decoder FC: pred = h2_cur @ fcW^T + fcb ; feed back
    if (dec) {
      if (hblk == 0) {
        const int o = tid & 1;
        const float* wrow = fcW + o * kH;
#pragma unroll
        for (int half = 0; half < 2; ++half) {
          const int bq = r0 + (tid >> 1) + half * 128;
          const __bf16* hrow = h2w + (size_t)bq * kH;
          float sum = fcb[o];
          for (int k = 0; k < kH; k += 8) {
            v8bf hv = *(const v8bf*)(hrow + k);
#pragma unroll
            for (int q = 0; q < 8; ++q) sum += (float)hv[q] * wrow[k + q];
          }
          out[((size_t)bq * kHor + (st - kT)) * 2 + o] = sum;  // [B, HOR, 2]
          xnext[bq * 2 + o] = sum;
        }
      }
      grid_bar(cnt, gen);
    }
  }
}

extern "C" void kernel_launch(void* const* d_in, const int* in_sizes, int n_in,
                              void* d_out, int out_size, void* d_ws, size_t ws_size,
                              hipStream_t stream) {
  (void)in_sizes; (void)n_in; (void)out_size; (void)ws_size;
  const float* x     = (const float*)d_in[0];
  const float* eWih0 = (const float*)d_in[1];
  const float* eWhh0 = (const float*)d_in[2];
  const float* eBi0  = (const float*)d_in[3];
  const float* eBh0  = (const float*)d_in[4];
  const float* eWih1 = (const float*)d_in[5];
  const float* eWhh1 = (const float*)d_in[6];
  const float* eBi1  = (const float*)d_in[7];
  const float* eBh1  = (const float*)d_in[8];
  const float* dWih0 = (const float*)d_in[9];
  const float* dWhh0 = (const float*)d_in[10];
  const float* dBi0  = (const float*)d_in[11];
  const float* dBh0  = (const float*)d_in[12];
  const float* dWih1 = (const float*)d_in[13];
  const float* dWhh1 = (const float*)d_in[14];
  const float* dBi1  = (const float*)d_in[15];
  const float* dBh1  = (const float*)d_in[16];
  const float* fcW   = (const float*)d_in[17];
  const float* fcb   = (const float*)d_in[18];

  char* ws = (char*)d_ws;
  size_t off = 0;
  auto take = [&](size_t bytes) -> void* {
    void* p = ws + off;
    off = (off + bytes + 255) & ~(size_t)255;
    return p;
  };
  unsigned* cnt  = (unsigned*)take(256);
  __bf16* h1buf  = (__bf16*)take((size_t)2 * kB * kH * 2);
  __bf16* h2buf  = (__bf16*)take((size_t)2 * kB * kH * 2);
  float*  xnext  = (float*)take((size_t)kB * 2 * sizeof(float));
  const size_t zbytes = off;                         // zero: barrier + h state + dec_in0
  __bf16* Whh0e  = (__bf16*)take((size_t)kG * kH * 2);
  __bf16* Wcat1e = (__bf16*)take((size_t)kG * 2 * kH * 2);
  __bf16* Whh0d  = (__bf16*)take((size_t)kG * kH * 2);
  __bf16* Wcat1d = (__bf16*)take((size_t)kG * 2 * kH * 2);

  // Allow >64KB dynamic LDS for the weight-resident persistent kernel.
  hipFuncSetAttribute((const void*)seq2seq_fused,
                      hipFuncAttributeMaxDynamicSharedMemorySize, kLdsBytes);

  hipMemsetAsync(d_ws, 0, zbytes, stream);

  cvt_weights<<<(kG * kH + kThr - 1) / kThr, kThr, 0, stream>>>(
      eWhh0, eWih1, eWhh1, dWhh0, dWih1, dWhh1,
      Whh0e, Wcat1e, Whh0d, Wcat1d);

  seq2seq_fused<<<kNWG, kThr, kLdsBytes, stream>>>(
      x, eWih0, eBi0, eBh0, eBi1, eBh1,
      dWih0, dBi0, dBh0, dBi1, dBh1, fcW, fcb,
      Whh0e, Wcat1e, Whh0d, Wcat1d,
      h1buf, h2buf, xnext, cnt, (float*)d_out);
}
